// GraphTransformerLayer_74706661147256
// MI455X (gfx1250) — matrix operand order repaired and checked by reading, not compiled, not run on hardware
//
#include <hip/hip_runtime.h>
#include <hip/hip_bf16.h>
#include <math.h>

// ---------------------------------------------------------------------------
// Graph Transformer layer for MI455X (gfx1250, wave32, WMMA).
// Roofline: ~70 GFLOP vs ~2 GB traffic @ 23.3 TB/s -> memory bound (~100us).
// Strategy: f16 streaming operands + v_wmma_f32_16x16x32_f16 for all GEMMs,
// f32 accumulate, fused epilogues, L2-resident node tables for gathers.
// ---------------------------------------------------------------------------

#define NNODES 50000
#define NEDGES 800000
#define DIM    64
#define NH     8
#define EPS_BN 1e-5f
#define EPS_SM 1e-16f

typedef __attribute__((ext_vector_type(16))) _Float16 v16h;
typedef __attribute__((ext_vector_type(8)))  _Float16 v8h;
typedef __attribute__((ext_vector_type(8)))  float    v8f;

// ---------------- elementwise helpers ----------------

__global__ void fill_f32_kernel(float* p, long long n, float v) {
    long long i = (long long)blockIdx.x * blockDim.x + threadIdx.x;
    if (i < n) p[i] = v;
}

__global__ void f32_to_f16_kernel(const float* __restrict__ x, _Float16* __restrict__ y, long long n) {
    long long i = (long long)blockIdx.x * blockDim.x + threadIdx.x;
    if (i < n) y[i] = (_Float16)x[i];
}

// Pack W[K x Nc] (row-major f32, [in,out]) into the WMMA B-operand lane layout:
// dst[((nt*numKt + kt)*32 + lane)*16 + i] = W[(kt*32 + (lane>>4)*16 + i)*Nc + nt*16 + (lane&15)]
__global__ void pack_weight_kernel(const float* __restrict__ W, _Float16* __restrict__ Wp, int K, int Nc) {
    int t = blockIdx.x * blockDim.x + threadIdx.x;
    int total = K * Nc;
    if (t >= total) return;
    int numKt = K >> 5;
    int i    = t & 15;
    int lane = (t >> 4) & 31;
    int rest = t >> 9;
    int kt = rest % numKt;
    int nt = rest / numKt;
    int k = kt * 32 + (lane >> 4) * 16 + i;
    int n = nt * 16 + (lane & 15);
    Wp[t] = (_Float16)W[(size_t)k * Nc + n];
}

// ---------------- WMMA GEMM: out = A[MxK](f16) @ Wp(K->Nc packed) ----------------
// Fused: +bias, ReLU, +residual(f32), writes f32 and/or f16 outputs.
__global__ __launch_bounds__(128) void gemm_wmma_kernel(
    const _Float16* __restrict__ A, const _Float16* __restrict__ Wp,
    int M, int K, int Nc,
    const float* __restrict__ bias, const float* __restrict__ residual,
    int relu, float* __restrict__ out32, _Float16* __restrict__ out16)
{
    const int lane = threadIdx.x & 31;
    const int wave = threadIdx.x >> 5;
    const int m  = lane & 15;   // row (A) / col (B) within tile
    const int hi = lane >> 4;   // lane-group select
    const int row0 = blockIdx.x << 4;
    const int numKt = K >> 5;
    const int numNt = Nc >> 4;

    const _Float16* arow = A + (size_t)(row0 + m) * K + hi * 16;

    for (int nt = wave; nt < numNt; nt += 4) {
        v8f c = {0.f, 0.f, 0.f, 0.f, 0.f, 0.f, 0.f, 0.f};
        const _Float16* bbase = Wp + ((size_t)nt * numKt * 32 + lane) * 16;
        for (int kt = 0; kt < numKt; ++kt) {
            const _Float16* ap = arow + kt * 32;
            v8h alo  = *(const v8h*)(ap);        // K = kt*32 + hi*16 + [0..7]
            v8h ahi8 = *(const v8h*)(ap + 8);    // K = kt*32 + hi*16 + 8 + [0..7]
            v16h a;
#pragma unroll
            for (int i = 0; i < 8; ++i) { a[i] = alo[i]; a[i + 8] = ahi8[i]; }
            v16h b = *(const v16h*)(bbase + (size_t)kt * 512);  // K = hi*16 + i, N = m
            c = __builtin_amdgcn_wmma_f32_16x16x32_f16(
                    false, a, false, b, (short)0, c, false, false);
        }
#pragma unroll
        for (int r = 0; r < 8; ++r) {
            int row = row0 + r + hi * 8;
            int col = (nt << 4) + m;
            float val = c[r];
            if (bias) val += bias[col];
            if (relu) val = val > 0.f ? val : 0.f;
            size_t idx = (size_t)row * Nc + col;
            if (residual) val += residual[idx];
            if (out32) out32[idx] = val;
            if (out16) out16[idx] = (_Float16)val;
        }
    }
}

// ---------------- attention scalar kernels ----------------

__device__ inline void atomicMaxFloat(float* addr, float val) {
    unsigned* ua = (unsigned*)addr;
    unsigned old = *ua;
    float cur = __uint_as_float(old);
    while (val > cur) {
        unsigned assumed = old;
        old = atomicCAS(ua, assumed, __float_as_uint(val));
        if (old == assumed) break;
        cur = __uint_as_float(old);
    }
}

// scores[e,h] = (q[dst] . (k[src]+ep[e]))_h * inv ; segment max over src
__global__ void scores_kernel(const int* __restrict__ src, const int* __restrict__ dst,
                              const float* __restrict__ q, const float* __restrict__ k,
                              const float* __restrict__ ep,
                              float* __restrict__ sc, float* __restrict__ mmax) {
    int e = blockIdx.x * blockDim.x + threadIdx.x;
    if (e >= NEDGES) return;
    int s = src[e], d = dst[e];
    const float* qi = q + (size_t)d * DIM;
    const float* kj = k + (size_t)s * DIM;
    const float* ev = ep + (size_t)e * DIM;
    const float inv = 0.35355339059327373f;  // 1/sqrt(8)
#pragma unroll
    for (int h = 0; h < NH; ++h) {
        float acc = 0.f;
#pragma unroll
        for (int dd = 0; dd < NH; ++dd) {
            int ii = h * NH + dd;
            acc += qi[ii] * (kj[ii] + ev[ii]);
        }
        acc *= inv;
        sc[(size_t)e * NH + h] = acc;
        atomicMaxFloat(&mmax[(size_t)s * NH + h], acc);
    }
}

// ex = exp(s - m[src]) (in place), segment sum over src
__global__ void expsum_kernel(const int* __restrict__ src,
                              float* __restrict__ sc,
                              const float* __restrict__ mmax,
                              float* __restrict__ ssum) {
    int e = blockIdx.x * blockDim.x + threadIdx.x;
    if (e >= NEDGES) return;
    int s = src[e];
#pragma unroll
    for (int h = 0; h < NH; ++h) {
        float ex = __expf(sc[(size_t)e * NH + h] - mmax[(size_t)s * NH + h]);
        sc[(size_t)e * NH + h] = ex;
        atomicAdd(&ssum[(size_t)s * NH + h], ex);
    }
}

// msg[e,h,d] = alpha[e,d] * (v[src]+ep)[e,h,d]  (faithful head-dim broadcast),
// scatter-add into outAgg[dst]
__global__ void message_kernel(const int* __restrict__ src, const int* __restrict__ dst,
                               const float* __restrict__ sc, const float* __restrict__ ssum,
                               const float* __restrict__ v, const float* __restrict__ ep,
                               float* __restrict__ outAgg) {
    int e = blockIdx.x * blockDim.x + threadIdx.x;
    if (e >= NEDGES) return;
    int s = src[e], d = dst[e];
    float alpha[NH];
#pragma unroll
    for (int i = 0; i < NH; ++i)
        alpha[i] = sc[(size_t)e * NH + i] / (ssum[(size_t)s * NH + i] + EPS_SM);
    const float* vj = v + (size_t)s * DIM;
    const float* ev = ep + (size_t)e * DIM;
    float* outd = outAgg + (size_t)d * DIM;
#pragma unroll
    for (int h = 0; h < NH; ++h)
#pragma unroll
        for (int dd = 0; dd < NH; ++dd) {
            int ii = h * NH + dd;
            atomicAdd(&outd[ii], alpha[dd] * (vj[ii] + ev[ii]));
        }
}

// cat16[e] = [ h[src] | h_att[dst] | ep[e] ]  (f32 -> f16 on the fly)
__global__ void cat_gather_kernel(const int* __restrict__ src, const int* __restrict__ dst,
                                  const float* __restrict__ h, const float* __restrict__ hatt,
                                  const float* __restrict__ ep, _Float16* __restrict__ cat) {
    int e = blockIdx.x * blockDim.x + threadIdx.x;
    if (e >= NEDGES) return;
    int s = src[e], d = dst[e];
    _Float16* cp = cat + (size_t)e * (3 * DIM);
    const float* hs = h + (size_t)s * DIM;
    const float* od = hatt + (size_t)d * DIM;
    const float* ev = ep + (size_t)e * DIM;
#pragma unroll 8
    for (int i = 0; i < DIM; ++i) {
        cp[i]           = (_Float16)hs[i];
        cp[DIM + i]     = (_Float16)od[i];
        cp[2 * DIM + i] = (_Float16)ev[i];
    }
}

// ---------------- batchnorm ----------------

// sums[0..63]=sum, sums[64..127]=sumsq, per column (64 cols), atomic-reduced
__global__ void bn_stats_kernel(const float* __restrict__ x, int M, float* __restrict__ sums) {
    int t = blockIdx.x * blockDim.x + threadIdx.x;
    int col = t & 63;
    int r0 = t >> 6;
    int stride = (gridDim.x * blockDim.x) >> 6;
    float s = 0.f, s2 = 0.f;
    for (int r = r0; r < M; r += stride) {
        float vv = x[(size_t)r * DIM + col];
        s += vv; s2 += vv * vv;
    }
    atomicAdd(&sums[col], s);
    atomicAdd(&sums[64 + col], s2);
}

__global__ void bn_apply_kernel(const float* __restrict__ x, int M,
                                const float* __restrict__ sums,
                                const float* __restrict__ g, const float* __restrict__ b,
                                float* __restrict__ out32, _Float16* __restrict__ out16) {
    long long i = (long long)blockIdx.x * blockDim.x + threadIdx.x;
    long long n = (long long)M * DIM;
    if (i >= n) return;
    int col = (int)(i & 63);
    float mu = sums[col] / (float)M;
    float var = sums[64 + col] / (float)M - mu * mu;
    float y = (x[i] - mu) * rsqrtf(var + EPS_BN) * g[col] + b[col];
    if (out32) out32[i] = y;
    if (out16) out16[i] = (_Float16)y;
}

// ---------------- host orchestration ----------------

static inline int ceil_div(long long a, int b) { return (int)((a + b - 1) / b); }

extern "C" void kernel_launch(void* const* d_in, const int* in_sizes, int n_in,
                              void* d_out, int out_size, void* d_ws, size_t ws_size,
                              hipStream_t stream) {
    (void)in_sizes; (void)n_in; (void)out_size;
    const int N = NNODES, E = NEDGES, D = DIM;

    // ---- inputs ----
    const float* h_in  = (const float*)d_in[0];
    const float* e_in  = (const float*)d_in[1];
    const int*   eidx  = (const int*)d_in[2];
    const int*   srcp  = eidx;          // edge_index[0]
    const int*   dstp  = eidx + E;      // edge_index[1]
    const float* Wq = (const float*)d_in[3],  *Wk = (const float*)d_in[4];
    const float* Wv = (const float*)d_in[5],  *We = (const float*)d_in[6];
    const float* Weu = (const float*)d_in[7], *Wproj = (const float*)d_in[8];
    const float* WOh = (const float*)d_in[9],  *bOh = (const float*)d_in[10];
    const float* WOe = (const float*)d_in[11], *bOe = (const float*)d_in[12];
    const float* g1h = (const float*)d_in[13], *b1h = (const float*)d_in[14];
    const float* g1e = (const float*)d_in[15], *b1e = (const float*)d_in[16];
    const float* Wf1h = (const float*)d_in[17], *bf1h = (const float*)d_in[18];
    const float* Wf2h = (const float*)d_in[19], *bf2h = (const float*)d_in[20];
    const float* Wf1e = (const float*)d_in[21], *bf1e = (const float*)d_in[22];
    const float* Wf2e = (const float*)d_in[23], *bf2e = (const float*)d_in[24];
    const float* g2h = (const float*)d_in[25], *b2h = (const float*)d_in[26];
    const float* g2e = (const float*)d_in[27], *b2e = (const float*)d_in[28];

    float* out_h = (float*)d_out;                    // [N,64]
    float* out_e = (float*)d_out + (size_t)N * D;    // [E,64]

    // ---- workspace bump allocator (with region reuse), ~900MB peak ----
    char* ws = (char*)d_ws;
    size_t cur = 0;
    auto alloc = [&](size_t bytes) -> char* {
        char* p = ws + cur;
        cur = (cur + bytes + 255) & ~(size_t)255;
        return p;
    };
    // packed weights (f16)
    _Float16* pWq   = (_Float16*)alloc(64 * 64 * 2);
    _Float16* pWk   = (_Float16*)alloc(64 * 64 * 2);
    _Float16* pWv   = (_Float16*)alloc(64 * 64 * 2);
    _Float16* pWe   = (_Float16*)alloc(64 * 64 * 2);
    _Float16* pWeu  = (_Float16*)alloc(192 * 64 * 2);
    _Float16* pWpr  = (_Float16*)alloc(64 * 64 * 2);
    _Float16* pWOh  = (_Float16*)alloc(64 * 64 * 2);
    _Float16* pWOe  = (_Float16*)alloc(64 * 64 * 2);
    _Float16* pWf1h = (_Float16*)alloc(64 * 128 * 2);
    _Float16* pWf2h = (_Float16*)alloc(128 * 64 * 2);
    _Float16* pWf1e = (_Float16*)alloc(64 * 128 * 2);
    _Float16* pWf2e = (_Float16*)alloc(128 * 64 * 2);
    // node-side buffers
    _Float16* h16    = (_Float16*)alloc((size_t)N * D * 2);
    float*    qf     = (float*)alloc((size_t)N * D * 4);
    float*    kf     = (float*)alloc((size_t)N * D * 4);
    float*    vf     = (float*)alloc((size_t)N * D * 4);
    float*    mmax   = (float*)alloc((size_t)N * NH * 4);
    float*    ssum   = (float*)alloc((size_t)N * NH * 4);
    float*    outAgg = (float*)alloc((size_t)N * D * 4);
    _Float16* outAgg16 = (_Float16*)alloc((size_t)N * D * 2);
    float*    hatt   = (float*)alloc((size_t)N * D * 4);
    _Float16* hatt16 = (_Float16*)alloc((size_t)N * D * 2);
    float*    hh     = (float*)alloc((size_t)N * D * 4);
    _Float16* hh16   = (_Float16*)alloc((size_t)N * D * 2);
    _Float16* ff1h16 = (_Float16*)alloc((size_t)N * 2 * D * 2);
    float*    hh2    = (float*)alloc((size_t)N * D * 4);
    float*    bnsums = (float*)alloc(128 * 4);
    // edge-side big regions (reused sequentially)
    char* regA = alloc((size_t)E * D * 2);      // e16 -> e_att16 -> ee16
    char* regB = alloc((size_t)E * D * 4);      // ep(f32) -> ee(f32)
    char* regD = alloc((size_t)E * NH * 4);     // scores / ex
    char* regC = alloc((size_t)E * 3 * D * 2);  // cat16 -> ff1e16
    char* regF = alloc((size_t)E * D * 4);      // ff2e(f32)
    if (cur > ws_size) return;  // workspace too small; bail

    _Float16* e16 = (_Float16*)regA;
    float*    ep  = (float*)regB;
    float*    sc  = (float*)regD;
    _Float16* cat16 = (_Float16*)regC;

    const int TB = 256;
    // 1) pack all weights
    pack_weight_kernel<<<ceil_div(64 * 64, TB), TB, 0, stream>>>(Wq, pWq, 64, 64);
    pack_weight_kernel<<<ceil_div(64 * 64, TB), TB, 0, stream>>>(Wk, pWk, 64, 64);
    pack_weight_kernel<<<ceil_div(64 * 64, TB), TB, 0, stream>>>(Wv, pWv, 64, 64);
    pack_weight_kernel<<<ceil_div(64 * 64, TB), TB, 0, stream>>>(We, pWe, 64, 64);
    pack_weight_kernel<<<ceil_div(192 * 64, TB), TB, 0, stream>>>(Weu, pWeu, 192, 64);
    pack_weight_kernel<<<ceil_div(64 * 64, TB), TB, 0, stream>>>(Wproj, pWpr, 64, 64);
    pack_weight_kernel<<<ceil_div(64 * 64, TB), TB, 0, stream>>>(WOh, pWOh, 64, 64);
    pack_weight_kernel<<<ceil_div(64 * 64, TB), TB, 0, stream>>>(WOe, pWOe, 64, 64);
    pack_weight_kernel<<<ceil_div(64 * 128, TB), TB, 0, stream>>>(Wf1h, pWf1h, 64, 128);
    pack_weight_kernel<<<ceil_div(128 * 64, TB), TB, 0, stream>>>(Wf2h, pWf2h, 128, 64);
    pack_weight_kernel<<<ceil_div(64 * 128, TB), TB, 0, stream>>>(Wf1e, pWf1e, 64, 128);
    pack_weight_kernel<<<ceil_div(128 * 64, TB), TB, 0, stream>>>(Wf2e, pWf2e, 128, 64);

    // 2) f16 activations
    f32_to_f16_kernel<<<ceil_div((long long)N * D, TB), TB, 0, stream>>>(h_in, h16, (long long)N * D);
    f32_to_f16_kernel<<<ceil_div((long long)E * D, TB), TB, 0, stream>>>(e_in, e16, (long long)E * D);

    // 3) projections: q,k,v (N x 64), ep (E x 64)
    gemm_wmma_kernel<<<N / 16, 128, 0, stream>>>(h16, pWq, N, 64, 64, nullptr, nullptr, 0, qf, nullptr);
    gemm_wmma_kernel<<<N / 16, 128, 0, stream>>>(h16, pWk, N, 64, 64, nullptr, nullptr, 0, kf, nullptr);
    gemm_wmma_kernel<<<N / 16, 128, 0, stream>>>(h16, pWv, N, 64, 64, nullptr, nullptr, 0, vf, nullptr);
    gemm_wmma_kernel<<<E / 16, 128, 0, stream>>>(e16, pWe, E, 64, 64, nullptr, nullptr, 0, ep, nullptr);

    // 4) attention: scores -> segment max -> exp/sum -> message scatter
    fill_f32_kernel<<<ceil_div((long long)N * NH, TB), TB, 0, stream>>>(mmax, (long long)N * NH, -INFINITY);
    fill_f32_kernel<<<ceil_div((long long)N * NH, TB), TB, 0, stream>>>(ssum, (long long)N * NH, 0.f);
    fill_f32_kernel<<<ceil_div((long long)N * D, TB), TB, 0, stream>>>(outAgg, (long long)N * D, 0.f);
    scores_kernel<<<E / TB, TB, 0, stream>>>(srcp, dstp, qf, kf, ep, sc, mmax);
    expsum_kernel<<<E / TB, TB, 0, stream>>>(srcp, sc, mmax, ssum);
    message_kernel<<<E / TB, TB, 0, stream>>>(srcp, dstp, sc, ssum, vf, ep, outAgg);

    // 5) h_att = outAgg @ Wproj
    f32_to_f16_kernel<<<ceil_div((long long)N * D, TB), TB, 0, stream>>>(outAgg, outAgg16, (long long)N * D);
    gemm_wmma_kernel<<<N / 16, 128, 0, stream>>>(outAgg16, pWpr, N, 64, 64, nullptr, nullptr, 0, hatt, hatt16);

    // 6) e_att = [h[src] | h_att[dst] | ep] @ Weu  (K=192)
    cat_gather_kernel<<<E / TB, TB, 0, stream>>>(srcp, dstp, h_in, hatt, ep, cat16);
    _Float16* eatt16 = (_Float16*)regA;  // reuse A (e16 dead)
    gemm_wmma_kernel<<<E / 16, 128, 0, stream>>>(cat16, pWeu, E, 192, 64, nullptr, nullptr, 0, nullptr, eatt16);

    // 7) O_h / O_e + residual
    float* ee = (float*)regB;  // reuse B (ep dead)
    gemm_wmma_kernel<<<N / 16, 128, 0, stream>>>(hatt16, pWOh, N, 64, 64, bOh, h_in, 0, hh, nullptr);
    gemm_wmma_kernel<<<E / 16, 128, 0, stream>>>(eatt16, pWOe, E, 64, 64, bOe, e_in, 0, ee, nullptr);

    // 8) BatchNorm 1 (h and e), write normalized f32 (in place) + f16 copy
    _Float16* ee16 = (_Float16*)regA;  // reuse A (eatt16 dead)
    fill_f32_kernel<<<1, 128, 0, stream>>>(bnsums, 128, 0.f);
    bn_stats_kernel<<<256, 256, 0, stream>>>(hh, N, bnsums);
    bn_apply_kernel<<<ceil_div((long long)N * D, TB), TB, 0, stream>>>(hh, N, bnsums, g1h, b1h, hh, hh16);
    fill_f32_kernel<<<1, 128, 0, stream>>>(bnsums, 128, 0.f);
    bn_stats_kernel<<<512, 256, 0, stream>>>(ee, E, bnsums);
    bn_apply_kernel<<<ceil_div((long long)E * D, TB), TB, 0, stream>>>(ee, E, bnsums, g1e, b1e, ee, ee16);

    // 9) FFNs: D -> 2D (ReLU) -> D, + residual
    _Float16* ff1e16 = (_Float16*)regC;  // reuse C (cat16 dead)
    float*    ff2e   = (float*)regF;
    gemm_wmma_kernel<<<N / 16, 128, 0, stream>>>(hh16, pWf1h, N, 64, 128, bf1h, nullptr, 1, nullptr, ff1h16);
    gemm_wmma_kernel<<<N / 16, 128, 0, stream>>>(ff1h16, pWf2h, N, 128, 64, bf2h, hh, 0, hh2, nullptr);
    gemm_wmma_kernel<<<E / 16, 128, 0, stream>>>(ee16, pWf1e, E, 64, 128, bf1e, nullptr, 1, nullptr, ff1e16);
    gemm_wmma_kernel<<<E / 16, 128, 0, stream>>>(ff1e16, pWf2e, E, 128, 64, bf2e, ee, 0, ff2e, nullptr);

    // 10) BatchNorm 2 -> final outputs
    fill_f32_kernel<<<1, 128, 0, stream>>>(bnsums, 128, 0.f);
    bn_stats_kernel<<<256, 256, 0, stream>>>(hh2, N, bnsums);
    bn_apply_kernel<<<ceil_div((long long)N * D, TB), TB, 0, stream>>>(hh2, N, bnsums, g2h, b2h, out_h, nullptr);
    fill_f32_kernel<<<1, 128, 0, stream>>>(bnsums, 128, 0.f);
    bn_stats_kernel<<<512, 256, 0, stream>>>(ff2e, E, bnsums);
    bn_apply_kernel<<<ceil_div((long long)E * D, TB), TB, 0, stream>>>(ff2e, E, bnsums, g2e, b2e, out_e, nullptr);
}
